// pw_Cheb_3p_uw_9835475107897
// MI455X (gfx1250) — compile-verified, matrix operand order checked by reading, not run on hardware
//
#include <hip/hip_runtime.h>

#define NN     20000
#define EE     320000
#define INF_   2048
#define OMASK  512
#define NCONN_ 16384
#define CL1_   256
#define CL2_   128
#define GG     32
#define OUTF_  3

typedef __attribute__((ext_vector_type(16))) __bf16 v16bf;
typedef __attribute__((ext_vector_type(8)))  float  v8f;

// ---------- helpers ----------
__device__ __forceinline__ unsigned int pack_bf16(float lo, float hi) {
  unsigned int ul = __float_as_uint(lo), uh = __float_as_uint(hi);
  ul = ul + 0x7FFFu + ((ul >> 16) & 1u);
  uh = uh + 0x7FFFu + ((uh >> 16) & 1u);
  return (ul >> 16) | (uh & 0xFFFF0000u);
}
__device__ __forceinline__ unsigned short f2bf(float f) {
  unsigned int u = __float_as_uint(f);
  u = u + 0x7FFFu + ((u >> 16) & 1u);
  return (unsigned short)(u >> 16);
}
__device__ __forceinline__ void atomicMaxF(float* addr, float value) {
  unsigned int* a = (unsigned int*)addr;
  unsigned int old = *a, assumed;
  do {
    assumed = old;
    if (__uint_as_float(assumed) >= value) break;
    old = atomicCAS(a, assumed, __float_as_uint(value));
  } while (old != assumed);
}

// ---------- tiny utility kernels ----------
__global__ void fill_kernel(float* p, float v, long long n) {
  long long i = (long long)blockIdx.x * blockDim.x + threadIdx.x;
  if (i < n) p[i] = v;
}
__global__ void scatter_w_kernel(const int* __restrict__ sm, const float* __restrict__ wv,
                                 float* __restrict__ Wt, int nconn, int S) {
  int i = blockIdx.x * blockDim.x + threadIdx.x;
  if (i < nconn) {
    int src = sm[2 * i], dst = sm[2 * i + 1];
    atomicAdd(&Wt[(size_t)dst * S + src], wv[i]);   // Bt layout [dst][src]
  }
}
__global__ void cvt_bf16_kernel(const float* __restrict__ src, unsigned short* __restrict__ dst,
                                long long n) {
  long long i = (long long)blockIdx.x * blockDim.x + threadIdx.x;
  if (i < n) dst[i] = f2bf(src[i]);
}
// src [Kd, Nc] row-major f32  ->  dst [Nc, Kd] bf16
__global__ void transpose_cvt_kernel(const float* __restrict__ src, unsigned short* __restrict__ dst,
                                     int Kd, int Nc) {
  int i = blockIdx.x * blockDim.x + threadIdx.x;
  if (i < Kd * Nc) {
    int k = i / Nc, n = i % Nc;
    dst[(size_t)n * Kd + k] = f2bf(src[i]);
  }
}

// ---------- graph normalization ----------
__global__ void deg_kernel(const int* __restrict__ row, const int* __restrict__ col,
                           float* __restrict__ deg, int E) {
  int e = blockIdx.x * blockDim.x + threadIdx.x;
  if (e < E && row[e] != col[e]) atomicAdd(&deg[row[e]], 1.0f);
}
__global__ void dis_kernel(const float* __restrict__ deg, float* __restrict__ dis, int n) {
  int i = blockIdx.x * blockDim.x + threadIdx.x;
  if (i < n) dis[i] = deg[i] > 0.0f ? rsqrtf(fmaxf(deg[i], 1.0f)) : 0.0f;
}
__global__ void norm_kernel(const int* __restrict__ row, const int* __restrict__ col,
                            const float* __restrict__ dis, float* __restrict__ nrm, int E) {
  int e = blockIdx.x * blockDim.x + threadIdx.x;
  if (e < E) nrm[e] = (row[e] != col[e]) ? -dis[row[e]] * dis[col[e]] : 0.0f;
}

// ---------- SPMM scatter: out[col] += norm * v[row] ----------
__global__ void spmm_kernel(const int* __restrict__ row, const int* __restrict__ col,
                            const float* __restrict__ nrm, const float* __restrict__ v,
                            float* __restrict__ out, int F) {
  int e = blockIdx.x;
  float w = nrm[e];
  if (w == 0.0f) return;
  const float* vs = v + (size_t)row[e] * F;
  float* od = out + (size_t)col[e] * F;
  for (int f = threadIdx.x; f < F; f += blockDim.x)
    atomicAdd(&od[f], w * vs[f]);
}

// a = 2*a - b  (tx2 = 2*spmm(tx1) - tx0)
__global__ void axpby_kernel(float* __restrict__ a, const float* __restrict__ b, long long n) {
  long long i = (long long)blockIdx.x * blockDim.x + threadIdx.x;
  if (i < n) a[i] = 2.0f * a[i] - b[i];
}

// ---------- WMMA bf16 GEMM ----------
// C[M,Nc](f32) (+)= A[M,Kd](f32, lda) @ Bt[Nc,Kd](bf16) + bias
// bias must always be a valid [Nc] pointer (use a zero vector when no bias).
// Block: 256 thr = 8 waves laid out 4(M) x 2(N). Wave tile: 16 rows x 32 cols
// (two 16x16 WMMA N-tiles sharing one A fragment). Block tile: 64 x 64.
__global__ __launch_bounds__(256) void wmma_gemm_bf16(
    const float* __restrict__ A, int lda,
    const unsigned short* __restrict__ Bt,
    const float* __restrict__ bias,
    float* __restrict__ C,
    int M, int Kd, int Nc, int accumulate) {
  const int lane  = threadIdx.x & 31;
  const int wave  = threadIdx.x >> 5;
  const int waveM = wave >> 1;                  // 0..3
  const int waveN = wave & 1;                   // 0..1
  const int rowBase = blockIdx.x * 64 + waveM * 16;
  const int colBase = blockIdx.y * 64 + waveN * 32;
  const bool fullTile = (rowBase + 16) <= M;    // wave-uniform

  const int l16   = lane & 15;
  const int hi    = lane >> 4;                  // half-wave select
  const int kbase = hi * 8;

  // A row this lane feeds; clamp instead of predicating loads (OOB rows are
  // computed but never stored: WMMA rows are independent).
  const int rowA  = min(rowBase + l16, M - 1);
  const int colB0 = colBase + l16;              // first N-tile column
  const int colB1 = colBase + 16 + l16;         // second N-tile column

  v8f acc0 = {}, acc1 = {};
  const int crow0 = rowBase + hi * 8;           // C/D layout: vgpr r -> row crow0+r
  if (accumulate) {
    if (fullTile) {
#pragma unroll
      for (int r = 0; r < 8; ++r) {
        const size_t rr = (size_t)(crow0 + r) * Nc;
        acc0[r] = C[rr + colB0];
        acc1[r] = C[rr + colB1];
      }
    } else {
#pragma unroll
      for (int r = 0; r < 8; ++r) {
        const size_t rr = (size_t)min(crow0 + r, M - 1) * Nc;
        acc0[r] = C[rr + colB0];
        acc1[r] = C[rr + colB1];
      }
    }
  }

  const float4* Arow = (const float4*)(A + (size_t)rowA * lda);
  const uint4*  Br0  = (const uint4*)(Bt + (size_t)colB0 * Kd);
  const uint4*  Br1  = (const uint4*)(Bt + (size_t)colB1 * Kd);

  for (int kk = 0; kk < Kd; kk += 32) {
    const int ia = (kk + kbase) >> 2;           // float4 index of K run 0
    const int ib = (kk + 16 + kbase) >> 2;      // float4 index of K run 1
    float4 x0 = Arow[ia];
    float4 x1 = Arow[ia + 1];
    float4 y0 = Arow[ib];
    float4 y1 = Arow[ib + 1];
    __builtin_prefetch((const float*)Arow + kk + 32, 0, 1);   // next K tile

    union { unsigned int u[8]; v16bf v; } a, b0, b1;
    a.u[0] = pack_bf16(x0.x, x0.y);  a.u[1] = pack_bf16(x0.z, x0.w);
    a.u[2] = pack_bf16(x1.x, x1.y);  a.u[3] = pack_bf16(x1.z, x1.w);
    a.u[4] = pack_bf16(y0.x, y0.y);  a.u[5] = pack_bf16(y0.z, y0.w);
    a.u[6] = pack_bf16(y1.x, y1.y);  a.u[7] = pack_bf16(y1.z, y1.w);

    const int ja = (kk + kbase) >> 3;           // uint4 index of 8-bf16 run 0
    const int jb = (kk + 16 + kbase) >> 3;      // uint4 index of 8-bf16 run 1
    uint4 p = Br0[ja], q = Br0[jb];
    b0.u[0] = p.x; b0.u[1] = p.y; b0.u[2] = p.z; b0.u[3] = p.w;
    b0.u[4] = q.x; b0.u[5] = q.y; b0.u[6] = q.z; b0.u[7] = q.w;
    p = Br1[ja]; q = Br1[jb];
    b1.u[0] = p.x; b1.u[1] = p.y; b1.u[2] = p.z; b1.u[3] = p.w;
    b1.u[4] = q.x; b1.u[5] = q.y; b1.u[6] = q.z; b1.u[7] = q.w;

    acc0 = __builtin_amdgcn_wmma_f32_16x16x32_bf16(false, a.v, false, b0.v,
                                                   (short)0, acc0, false, false);
    acc1 = __builtin_amdgcn_wmma_f32_16x16x32_bf16(false, a.v, false, b1.v,
                                                   (short)0, acc1, false, false);
  }

  const float bv0 = bias[colB0];
  const float bv1 = bias[colB1];
  if (fullTile) {
#pragma unroll
    for (int r = 0; r < 8; ++r) {
      const size_t rr = (size_t)(crow0 + r) * Nc;
      C[rr + colB0] = acc0[r] + bv0;
      C[rr + colB1] = acc1[r] + bv1;
    }
  } else {
#pragma unroll
    for (int r = 0; r < 8; ++r) {
      int rr = crow0 + r;
      if (rr < M) {
        C[(size_t)rr * Nc + colB0] = acc0[r] + bv0;
        C[(size_t)rr * Nc + colB1] = acc1[r] + bv1;
      }
    }
  }
}

// ---------- fused relu + batchnorm ----------
__global__ void bn_stats_kernel(const float* __restrict__ z, float* __restrict__ sum,
                                float* __restrict__ sumsq, int Nrows, int C, int rowsPerBlock) {
  int c = threadIdx.x;                        // blockDim.x == C
  int r0 = blockIdx.x * rowsPerBlock;
  int r1 = min(r0 + rowsPerBlock, Nrows);
  float s = 0.0f, q = 0.0f;
  for (int r = r0; r < r1; ++r) {
    float v = fmaxf(z[(size_t)r * C + c], 0.0f);
    s += v; q += v * v;
  }
  atomicAdd(&sum[c], s);
  atomicAdd(&sumsq[c], q);
}
__global__ void bn_apply_kernel(float* __restrict__ z, const float* __restrict__ sum,
                                const float* __restrict__ sumsq, const float* __restrict__ gamma,
                                const float* __restrict__ beta, int Nrows, int C) {
  long long i = (long long)blockIdx.x * blockDim.x + threadIdx.x;
  if (i >= (long long)Nrows * C) return;
  int c = (int)(i % C);
  float invN = 1.0f / (float)Nrows;
  float mu = sum[c] * invN;
  float var = sumsq[c] * invN - mu * mu;
  float v = fmaxf(z[i], 0.0f);
  z[i] = (v - mu) * rsqrtf(var + 1e-5f) * gamma[c] + beta[c];
}

// ---------- pooling ----------
__global__ void pool_kernel(const float* __restrict__ h, const int* __restrict__ batch,
                            float* __restrict__ s, float* __restrict__ cnt,
                            float* __restrict__ mx, int Nrows, int C, int rowsPerBlock) {
  int c = threadIdx.x;                        // blockDim.x == C
  int r0 = blockIdx.x * rowsPerBlock;
  int r1 = min(r0 + rowsPerBlock, Nrows);
  for (int r = r0; r < r1; ++r) {
    int g = batch[r];
    float v = h[(size_t)r * C + c];
    atomicAdd(&s[g * C + c], v);
    atomicMaxF(&mx[g * C + c], v);
    if (c == 0) atomicAdd(&cnt[g], 1.0f);
  }
}

// ---------- head: cat [G,3C] then lin [G,3] into d_out ----------
__global__ void final_kernel(const float* __restrict__ s, const float* __restrict__ cnt,
                             const float* __restrict__ mx, const float* __restrict__ w_lin,
                             const float* __restrict__ b_lin, float* __restrict__ out,
                             int C, int G) {
  int t = threadIdx.x;                        // 3*C threads
  float* cat = out;
  float* lin = out + (size_t)G * 3 * C;
  for (int g = 0; g < G; ++g) {
    float cn = cnt[g];
    float v;
    if (t < C) {
      v = s[g * C + t];
    } else if (t < 2 * C) {
      v = s[g * C + (t - C)] / fmaxf(cn, 1.0f);
    } else {
      float m = mx[g * C + (t - 2 * C)];
      v = (m > -1.0e37f) ? m : 0.0f;          // finite guard for empty segments
    }
    cat[(size_t)g * 3 * C + t] = v;
  }
  __syncthreads();
  if (t < G * OUTF_) {
    int g = t / OUTF_, o = t % OUTF_;
    float acc = b_lin[o];
    for (int j = 0; j < 3 * C; ++j) acc += cat[(size_t)g * 3 * C + j] * w_lin[j * OUTF_ + o];
    lin[g * OUTF_ + o] = acc;
  }
}

// ---------- orchestration ----------
extern "C" void kernel_launch(void* const* d_in, const int* in_sizes, int n_in,
                              void* d_out, int out_size, void* d_ws, size_t ws_size,
                              hipStream_t stream) {
  const float* x      = (const float*)d_in[0];
  const int*   edge   = (const int*)d_in[1];
  const int*   row    = edge;
  const int*   col    = edge + EE;
  const int*   batch  = (const int*)d_in[2];
  const int*   sm     = (const int*)d_in[3];
  const float* w_mask = (const float*)d_in[4];
  const float* b_mask = (const float*)d_in[5];
  const float* theta1 = (const float*)d_in[6];
  const float* bc1    = (const float*)d_in[7];
  const float* gamma1 = (const float*)d_in[8];
  const float* beta1  = (const float*)d_in[9];
  const float* theta2 = (const float*)d_in[10];
  const float* bc2    = (const float*)d_in[11];
  const float* gamma2 = (const float*)d_in[12];
  const float* beta2  = (const float*)d_in[13];
  const float* w_lin  = (const float*)d_in[14];
  const float* b_lin  = (const float*)d_in[15];
  float* out = (float*)d_out;

  char* ws = (char*)d_ws;
  size_t off = 0;
  auto carve = [&](size_t bytes) -> void* {
    void* p = ws + off;
    off += (bytes + 255) & ~(size_t)255;
    return p;
  };
  float*          Wt_f   = (float*)carve((size_t)OMASK * OMASK * 4);
  unsigned short* Wt_bf  = (unsigned short*)carve((size_t)OMASK * OMASK * 2);
  unsigned short* th1t   = (unsigned short*)carve((size_t)3 * CL1_ * OMASK * 2); // [256,512] each
  unsigned short* th2t   = (unsigned short*)carve((size_t)3 * CL2_ * CL1_ * 2);  // [128,256] each
  float*          H0     = (float*)carve((size_t)NN * OMASK * 4);
  float*          T1     = (float*)carve((size_t)NN * OMASK * 4);
  float*          T2     = (float*)carve((size_t)NN * OMASK * 4);
  float*          C1     = (float*)carve((size_t)NN * CL1_ * 4);
  float*          C2     = (float*)carve((size_t)NN * CL2_ * 4);
  float*          deg    = (float*)carve((size_t)NN * 4);
  float*          dis    = (float*)carve((size_t)NN * 4);
  float*          nrm    = (float*)carve((size_t)EE * 4);
  float*          st_sum = (float*)carve(CL1_ * 4);
  float*          st_sq  = (float*)carve(CL1_ * 4);
  float*          zeros  = (float*)carve(OMASK * 4);
  float*          p_s    = (float*)carve((size_t)GG * CL2_ * 4);
  float*          p_cnt  = (float*)carve((size_t)GG * 4);
  float*          p_mx   = (float*)carve((size_t)GG * CL2_ * 4);

  auto blocks = [](long long n) { return (unsigned)((n + 255) / 256); };

  // --- zero bias vector (used when a GEMM has no bias term) ---
  fill_kernel<<<blocks(OMASK), 256, 0, stream>>>(zeros, 0.0f, OMASK);

  // --- weights: densify masked linear (already in Bt layout) + bf16 conversion ---
  fill_kernel<<<blocks((long long)OMASK * OMASK), 256, 0, stream>>>(Wt_f, 0.0f, (long long)OMASK * OMASK);
  scatter_w_kernel<<<blocks(NCONN_), 256, 0, stream>>>(sm, w_mask, Wt_f, NCONN_, OMASK);
  cvt_bf16_kernel<<<blocks((long long)OMASK * OMASK), 256, 0, stream>>>(Wt_f, Wt_bf, (long long)OMASK * OMASK);
  for (int k = 0; k < 3; ++k) {
    transpose_cvt_kernel<<<blocks(OMASK * CL1_), 256, 0, stream>>>(
        theta1 + (size_t)k * OMASK * CL1_, th1t + (size_t)k * CL1_ * OMASK, OMASK, CL1_);
    transpose_cvt_kernel<<<blocks(CL1_ * CL2_), 256, 0, stream>>>(
        theta2 + (size_t)k * CL1_ * CL2_, th2t + (size_t)k * CL2_ * CL1_, CL1_, CL2_);
  }

  // --- graph normalization coefficients ---
  fill_kernel<<<blocks(NN), 256, 0, stream>>>(deg, 0.0f, NN);
  deg_kernel<<<blocks(EE), 256, 0, stream>>>(row, col, deg, EE);
  dis_kernel<<<blocks(NN), 256, 0, stream>>>(deg, dis, NN);
  norm_kernel<<<blocks(EE), 256, 0, stream>>>(row, col, dis, nrm, EE);

  // --- sparse masked linear as dense bf16 WMMA GEMM: H0 = x[:, :512] @ W + b_mask ---
  dim3 g1((NN + 63) / 64, OMASK / 64);
  wmma_gemm_bf16<<<g1, 256, 0, stream>>>(x, INF_, Wt_bf, b_mask, H0, NN, OMASK, OMASK, 0);

  // --- Cheb layer 1 (K=3): C1 = T0@th0 + T1@th1 + T2@th2 + bc1 ---
  dim3 gc1((NN + 63) / 64, CL1_ / 64);
  fill_kernel<<<blocks((long long)NN * OMASK), 256, 0, stream>>>(T1, 0.0f, (long long)NN * OMASK);
  spmm_kernel<<<EE, 256, 0, stream>>>(row, col, nrm, H0, T1, OMASK);
  wmma_gemm_bf16<<<gc1, 256, 0, stream>>>(H0, OMASK, th1t + 0 * (size_t)CL1_ * OMASK, zeros, C1, NN, OMASK, CL1_, 0);
  wmma_gemm_bf16<<<gc1, 256, 0, stream>>>(T1, OMASK, th1t + 1 * (size_t)CL1_ * OMASK, zeros, C1, NN, OMASK, CL1_, 1);
  fill_kernel<<<blocks((long long)NN * OMASK), 256, 0, stream>>>(T2, 0.0f, (long long)NN * OMASK);
  spmm_kernel<<<EE, 256, 0, stream>>>(row, col, nrm, T1, T2, OMASK);
  axpby_kernel<<<blocks((long long)NN * OMASK), 256, 0, stream>>>(T2, H0, (long long)NN * OMASK);
  wmma_gemm_bf16<<<gc1, 256, 0, stream>>>(T2, OMASK, th1t + 2 * (size_t)CL1_ * OMASK, bc1, C1, NN, OMASK, CL1_, 1);

  // --- relu + BN1 (in place on C1) ---
  fill_kernel<<<1, 256, 0, stream>>>(st_sum, 0.0f, CL1_);
  fill_kernel<<<1, 256, 0, stream>>>(st_sq, 0.0f, CL1_);
  bn_stats_kernel<<<(NN + 63) / 64, CL1_, 0, stream>>>(C1, st_sum, st_sq, NN, CL1_, 64);
  bn_apply_kernel<<<blocks((long long)NN * CL1_), 256, 0, stream>>>(C1, st_sum, st_sq, gamma1, beta1, NN, CL1_);

  // --- Cheb layer 2 (K=3): C2 = C1@th0 + T1@th1 + T2@th2 + bc2 ---
  dim3 gc2((NN + 63) / 64, CL2_ / 64);
  fill_kernel<<<blocks((long long)NN * CL1_), 256, 0, stream>>>(T1, 0.0f, (long long)NN * CL1_);
  spmm_kernel<<<EE, 256, 0, stream>>>(row, col, nrm, C1, T1, CL1_);
  wmma_gemm_bf16<<<gc2, 256, 0, stream>>>(C1, CL1_, th2t + 0 * (size_t)CL2_ * CL1_, zeros, C2, NN, CL1_, CL2_, 0);
  wmma_gemm_bf16<<<gc2, 256, 0, stream>>>(T1, CL1_, th2t + 1 * (size_t)CL2_ * CL1_, zeros, C2, NN, CL1_, CL2_, 1);
  fill_kernel<<<blocks((long long)NN * CL1_), 256, 0, stream>>>(T2, 0.0f, (long long)NN * CL1_);
  spmm_kernel<<<EE, 256, 0, stream>>>(row, col, nrm, T1, T2, CL1_);
  axpby_kernel<<<blocks((long long)NN * CL1_), 256, 0, stream>>>(T2, C1, (long long)NN * CL1_);
  wmma_gemm_bf16<<<gc2, 256, 0, stream>>>(T2, CL1_, th2t + 2 * (size_t)CL2_ * CL1_, bc2, C2, NN, CL1_, CL2_, 1);

  // --- relu + BN2 (in place on C2) ---
  fill_kernel<<<1, 256, 0, stream>>>(st_sum, 0.0f, CL2_);
  fill_kernel<<<1, 256, 0, stream>>>(st_sq, 0.0f, CL2_);
  bn_stats_kernel<<<(NN + 63) / 64, CL2_, 0, stream>>>(C2, st_sum, st_sq, NN, CL2_, 64);
  bn_apply_kernel<<<blocks((long long)NN * CL2_), 256, 0, stream>>>(C2, st_sum, st_sq, gamma2, beta2, NN, CL2_);

  // --- pooling + head ---
  fill_kernel<<<blocks(GG * CL2_), 256, 0, stream>>>(p_s, 0.0f, GG * CL2_);
  fill_kernel<<<1, 32, 0, stream>>>(p_cnt, 0.0f, GG);
  fill_kernel<<<blocks(GG * CL2_), 256, 0, stream>>>(p_mx, -3.0e38f, GG * CL2_);
  pool_kernel<<<(NN + 63) / 64, CL2_, 0, stream>>>(C2, batch, p_s, p_cnt, p_mx, NN, CL2_, 64);
  final_kernel<<<1, 3 * CL2_, 0, stream>>>(p_s, p_cnt, p_mx, w_lin, b_lin, out, CL2_, GG);
}
